// DBCR_26156350833260
// MI455X (gfx1250) — matrix operand order repaired and checked by reading, not compiled, not run on hardware
//
#include <hip/hip_runtime.h>
#include <hip/hip_bf16.h>

// ---------------------------------------------------------------------------
// Problem constants (match reference)
// ---------------------------------------------------------------------------
#define N_U     100000
#define N_I     50000
#define N_B     10
#define DIM     64
#define NLAYER  2
#define N_EDGES 1000000
#define BATCH   2048
#define DROPOUT 0.25f
#define TEMP    0.2f
#define LAMBDA_1 0.2f
#define LAMBDA_2 0.2f
#define LAMBDA_3 1e-7f

// scalar accumulator slots (floats in workspace)
#define S_BPR   0
#define S_PMIN  1
#define S_PMAX  2
#define S_POS_U 3
#define S_POS_I 4
#define S_NEG_U 5
#define S_NEG_I 6
#define S_REG   7
#define S_BCLN  8
#define S_BCLP  9
#define NSCAL   16

typedef __attribute__((ext_vector_type(16))) __bf16        v16bf;
typedef __attribute__((ext_vector_type(8)))  float         v8f;
typedef __attribute__((ext_vector_type(8)))  unsigned int  v8u;

// ---------------------------------------------------------------------------
// small helpers
// ---------------------------------------------------------------------------
__device__ __forceinline__ unsigned hash_u32(unsigned x) {
    x ^= x >> 16; x *= 0x7feb352du;
    x ^= x >> 15; x *= 0x846ca68bu;
    x ^= x >> 16; return x;
}

__device__ __forceinline__ unsigned f2bf_u(float f) {
    unsigned u = __float_as_uint(f);
    u += 0x7fffu + ((u >> 16) & 1u);   // round-to-nearest-even
    return u >> 16;
}

__global__ void k_zero(float* p, int n) {
    int i = blockIdx.x * blockDim.x + threadIdx.x;
    if (i < n) p[i] = 0.0f;
}

__global__ void k_add(float* dst, const float* src, int n) {
    int i = blockIdx.x * blockDim.x + threadIdx.x;
    if (i < n) dst[i] += src[i];
}

// ---------------------------------------------------------------------------
// SpMM with fused inverted dropout: Edst[dstIdx[e]] += drop(vals[e]) * Esrc[srcIdx[e]]
// 16 threads per edge, each thread does a float4 slice (DIM = 64).
// ---------------------------------------------------------------------------
__global__ void k_spmm(const float* __restrict__ vals,
                       const int*   __restrict__ srcIdx,
                       const int*   __restrict__ dstIdx,
                       const float* __restrict__ Esrc,
                       float*       __restrict__ Edst,
                       unsigned seed, int nEdges)
{
    long long t = (long long)blockIdx.x * blockDim.x + threadIdx.x;
    int e   = (int)(t >> 4);
    int sub = (int)(t & 15);
    if (e >= nEdges) return;
    unsigned r = hash_u32((unsigned)e ^ (seed * 0x9E3779B9u));
    float keep = ((r >> 8) * (1.0f / 16777216.0f)) >= DROPOUT ? 1.0f : 0.0f;
    float v = keep * vals[e] * (1.0f / (1.0f - DROPOUT));
    if (v == 0.0f) return;
    const float4 x = *(const float4*)(Esrc + (long long)srcIdx[e] * DIM + sub * 4);
    float* d = Edst + (long long)dstIdx[e] * DIM + sub * 4;
    atomicAdd(d + 0, v * x.x);
    atomicAdd(d + 1, v * x.y);
    atomicAdd(d + 2, v * x.z);
    atomicAdd(d + 3, v * x.w);
}

// ---------------------------------------------------------------------------
// aug_vals[e] = sigmoid(dot(E_u[row], E_v[col])) * adj_vals[e]
// ---------------------------------------------------------------------------
__global__ void k_edge_logits(const float* __restrict__ Eu,
                              const float* __restrict__ Ev,
                              const float* __restrict__ vals,
                              const int*   __restrict__ rows,
                              const int*   __restrict__ cols,
                              float*       __restrict__ aug, int nEdges)
{
    int e = blockIdx.x * blockDim.x + threadIdx.x;
    if (e >= nEdges) return;
    const float4* a = (const float4*)(Eu + (long long)rows[e] * DIM);
    const float4* b = (const float4*)(Ev + (long long)cols[e] * DIM);
    float s = 0.0f;
#pragma unroll
    for (int k = 0; k < DIM / 4; ++k) {
        float4 x = a[k], y = b[k];
        s += x.x * y.x + x.y * y.y + x.z * y.z + x.w * y.w;
    }
    aug[e] = vals[e] / (1.0f + __expf(-s));
}

// ---------------------------------------------------------------------------
// bf16 conversions, packed 2 elements / dword (fold 1/TEMP scale for Z batch)
// ---------------------------------------------------------------------------
__global__ void k_conv_bf16(const float2* __restrict__ src,
                            unsigned int* __restrict__ dst, int n2)
{
    int i = blockIdx.x * blockDim.x + threadIdx.x;
    if (i < n2) {
        float2 v = src[i];
        dst[i] = f2bf_u(v.x) | (f2bf_u(v.y) << 16);
    }
}

__global__ void k_gather_bf16(const float* __restrict__ Z,
                              const int*   __restrict__ ids,
                              unsigned int* __restrict__ dst,
                              float scale, int batch)
{
    int i = blockIdx.x * blockDim.x + threadIdx.x;    // batch * 32 packed dwords
    if (i >= batch * (DIM / 2)) return;
    int b = i >> 5, d2 = i & 31;
    const float2 v = *(const float2*)(Z + (long long)ids[b] * DIM + d2 * 2);
    dst[i] = f2bf_u(v.x * scale) | (f2bf_u(v.y * scale) << 16);
}

// ---------------------------------------------------------------------------
// PCL negative term: negacc[b] = sum_j exp( dot(Zb[b], E[j]) )  (Zb pre-scaled 1/T)
// v_wmma_f32_16x16x32_bf16, K = 64 -> 2 WMMAs per 16x16 score tile.
// Pipeline manually unrolled x2 with disjoint ping/pong buffer sets: loads for
// stage k+1 issue into the other buffer set before the WMMAs of stage k, with
// no register rotation copies, so each WMMA pair only ever waits on loads that
// were issued one stage earlier.
// grid = (BATCH/16, colSlices), block = 256 (8 waves); each wave strides tiles.
// ---------------------------------------------------------------------------
__device__ __forceinline__ void load_btile(const unsigned int* __restrict__ Ebf,
                                           int tile, int n, int h,
                                           v8u& b0, v8u& b1)
{
    const unsigned int* brow = Ebf + (long long)(tile * 16 + n) * 32;
    const uint4 p0 = *(const uint4*)(brow + h * 8);
    const uint4 p1 = *(const uint4*)(brow + h * 8 + 4);
    b0[0] = p0.x; b0[1] = p0.y; b0[2] = p0.z; b0[3] = p0.w;
    b0[4] = p1.x; b0[5] = p1.y; b0[6] = p1.z; b0[7] = p1.w;
    const uint4 q0 = *(const uint4*)(brow + 16 + h * 8);
    const uint4 q1 = *(const uint4*)(brow + 16 + h * 8 + 4);
    b1[0] = q0.x; b1[1] = q0.y; b1[2] = q0.z; b1[3] = q0.w;
    b1[4] = q1.x; b1[5] = q1.y; b1[6] = q1.z; b1[7] = q1.w;
}

__global__ void k_pcl_neg_wmma(const unsigned int* __restrict__ Zbf,  // [BATCH][32] dwords
                               const unsigned int* __restrict__ Ebf,  // [N][32] dwords
                               float* __restrict__ negacc,            // [BATCH]
                               int N)
{
    const int lane = threadIdx.x & 31;
    const int wave = threadIdx.x >> 5;
    const int wpb  = blockDim.x >> 5;
    const int i0   = blockIdx.x * 16;          // batch-row tile
    const int h    = lane >> 4;                // half of the wave
    const int n    = lane & 15;                // column / row-in-tile index

    // ---- A tile: rows Zbf[i0 + n], contiguous k = 16h + s layout, two K-chunks ----
    v8u au0, au1;
    {
        const unsigned int* arow = Zbf + (long long)(i0 + n) * 32;
#pragma unroll
        for (int c = 0; c < 2; ++c) {
            const uint4 q0 = *(const uint4*)(arow + c * 16 + h * 8);
            const uint4 q1 = *(const uint4*)(arow + c * 16 + h * 8 + 4);
            v8u& au = c ? au1 : au0;
            au[0] = q0.x; au[1] = q0.y; au[2] = q0.z; au[3] = q0.w;
            au[4] = q1.x; au[5] = q1.y; au[6] = q1.z; au[7] = q1.w;
        }
    }
    const v16bf a0 = __builtin_bit_cast(v16bf, au0);
    const v16bf a1 = __builtin_bit_cast(v16bf, au1);

    float acc[8] = {0, 0, 0, 0, 0, 0, 0, 0};

    const int tiles  = N >> 4;
    const int stride = gridDim.y * wpb;

    v8u p0b0 = {}, p0b1 = {};        // ping buffer set
    v8u p1b0 = {}, p1b1 = {};        // pong buffer set

    int t = blockIdx.y * wpb + wave;
    bool more = t < tiles;           // wave-uniform
    if (more) load_btile(Ebf, t, n, h, p0b0, p0b1);

    while (more) {
        // ---- stage A: prefetch into pong, compute on ping ----
        const int t1 = t + stride;
        const bool m1 = t1 < tiles;
        if (m1) {
            load_btile(Ebf, t1, n, h, p1b0, p1b1);
            __builtin_prefetch(Ebf + (long long)((t1 + stride < tiles ? t1 + stride : t1) * 16 + n) * 32, 0, 1);
        }
        {
            v8f c = {};
            c = __builtin_amdgcn_wmma_f32_16x16x32_bf16(false, a0, false,
                    __builtin_bit_cast(v16bf, p0b0), (short)0, c, false, false);
            c = __builtin_amdgcn_wmma_f32_16x16x32_bf16(false, a1, false,
                    __builtin_bit_cast(v16bf, p0b1), (short)0, c, false, false);
#pragma unroll
            for (int r = 0; r < 8; ++r) acc[r] += __expf(c[r]);
        }
        if (!m1) break;

        // ---- stage B: prefetch into ping, compute on pong ----
        const int t2 = t1 + stride;
        const bool m2 = t2 < tiles;
        if (m2) {
            load_btile(Ebf, t2, n, h, p0b0, p0b1);
            __builtin_prefetch(Ebf + (long long)((t2 + stride < tiles ? t2 + stride : t2) * 16 + n) * 32, 0, 1);
        }
        {
            v8f d = {};
            d = __builtin_amdgcn_wmma_f32_16x16x32_bf16(false, a0, false,
                    __builtin_bit_cast(v16bf, p1b0), (short)0, d, false, false);
            d = __builtin_amdgcn_wmma_f32_16x16x32_bf16(false, a1, false,
                    __builtin_bit_cast(v16bf, p1b1), (short)0, d, false, false);
#pragma unroll
            for (int r = 0; r < 8; ++r) acc[r] += __expf(d[r]);
        }
        t = t2;
        more = m2;
    }

    // reduce over the 16 lanes of each half (same row set, disjoint columns)
#pragma unroll
    for (int m = 1; m < 16; m <<= 1) {
#pragma unroll
        for (int r = 0; r < 8; ++r) acc[r] += __shfl_xor(acc[r], m, 32);
    }
    if (n == 0) {
#pragma unroll
        for (int r = 0; r < 8; ++r)
            atomicAdd(&negacc[i0 + r + 8 * h], acc[r]);
    }
}

// ---------------------------------------------------------------------------
// BPR: pos/neg scores, softplus loss, store pos_scores
// ---------------------------------------------------------------------------
__global__ void k_bpr(const float* __restrict__ Eu, const float* __restrict__ Ev,
                      const int* __restrict__ uids, const int* __restrict__ pos,
                      const int* __restrict__ neg,
                      float* __restrict__ ps_out, float* __restrict__ scal, int batch)
{
    __shared__ float sm[256];
    int b = blockIdx.x * blockDim.x + threadIdx.x;
    float term = 0.0f;
    if (b < batch) {
        const float4* u = (const float4*)(Eu + (long long)uids[b] * DIM);
        const float4* p = (const float4*)(Ev + (long long)pos[b]  * DIM);
        const float4* q = (const float4*)(Ev + (long long)neg[b]  * DIM);
        float ps = 0.0f, ns = 0.0f;
#pragma unroll
        for (int k = 0; k < DIM / 4; ++k) {
            float4 x = u[k], y = p[k], z = q[k];
            ps += x.x * y.x + x.y * y.y + x.z * y.z + x.w * y.w;
            ns += x.x * z.x + x.y * z.y + x.z * z.z + x.w * z.w;
        }
        ps_out[b] = ps;
        float x = ps - ns;   // -log(sigmoid(x)) = softplus(-x), stable form:
        term = (x > 0.0f) ? __logf(1.0f + __expf(-x)) : (-x + __logf(1.0f + __expf(x)));
    }
    sm[threadIdx.x] = term;
    __syncthreads();
    for (int s = blockDim.x >> 1; s > 0; s >>= 1) {
        if (threadIdx.x < s) sm[threadIdx.x] += sm[threadIdx.x + s];
        __syncthreads();
    }
    if (threadIdx.x == 0) atomicAdd(&scal[S_BPR], sm[0]);
}

// single-block min/max over pos_scores
__global__ void k_minmax(const float* __restrict__ ps, float* __restrict__ scal, int batch)
{
    __shared__ float smin[256], smax[256];
    float mn = 3.4e38f, mx = -3.4e38f;
    for (int i = threadIdx.x; i < batch; i += blockDim.x) {
        float v = ps[i];
        mn = fminf(mn, v); mx = fmaxf(mx, v);
    }
    smin[threadIdx.x] = mn; smax[threadIdx.x] = mx;
    __syncthreads();
    for (int s = blockDim.x >> 1; s > 0; s >>= 1) {
        if (threadIdx.x < s) {
            smin[threadIdx.x] = fminf(smin[threadIdx.x], smin[threadIdx.x + s]);
            smax[threadIdx.x] = fmaxf(smax[threadIdx.x], smax[threadIdx.x + s]);
        }
        __syncthreads();
    }
    if (threadIdx.x == 0) { scal[S_PMIN] = smin[0]; scal[S_PMAX] = smax[0]; }
}

// ---------------------------------------------------------------------------
// PCL positive term: clip(dot(Z[id], E[id]) / T, -5, 5) summed into scal[slot]
// ---------------------------------------------------------------------------
__global__ void k_pcl_pos(const float* __restrict__ Z, const float* __restrict__ E,
                          const int* __restrict__ ids, float* __restrict__ scal,
                          int slot, int batch)
{
    __shared__ float sm[256];
    int b = blockIdx.x * blockDim.x + threadIdx.x;
    float term = 0.0f;
    if (b < batch) {
        const float4* z = (const float4*)(Z + (long long)ids[b] * DIM);
        const float4* e = (const float4*)(E + (long long)ids[b] * DIM);
        float s = 0.0f;
#pragma unroll
        for (int k = 0; k < DIM / 4; ++k) {
            float4 x = z[k], y = e[k];
            s += x.x * y.x + x.y * y.y + x.z * y.z + x.w * y.w;
        }
        term = fminf(fmaxf(s / TEMP, -5.0f), 5.0f);
    }
    sm[threadIdx.x] = term;
    __syncthreads();
    for (int s = blockDim.x >> 1; s > 0; s >>= 1) {
        if (threadIdx.x < s) sm[threadIdx.x] += sm[threadIdx.x + s];
        __syncthreads();
    }
    if (threadIdx.x == 0) atomicAdd(&scal[slot], sm[0]);
}

// ---------------------------------------------------------------------------
// BCL: bucketed contrastive loss
// ---------------------------------------------------------------------------
__global__ void k_bcl(const float* __restrict__ Eu, const float* __restrict__ Ev,
                      const float* __restrict__ Eb,
                      const int* __restrict__ uids, const int* __restrict__ pos,
                      const float* __restrict__ ps, float* __restrict__ scal, int batch)
{
    __shared__ float sn[256], sp[256];
    int b = blockIdx.x * blockDim.x + threadIdx.x;
    float tneg = 0.0f, tpos = 0.0f;
    if (b < batch) {
        const float mn = scal[S_PMIN], mx = scal[S_PMAX];
        float w = (ps[b] - mn) / (mx - mn + 1e-9f);
        int rel = min(max((int)(w * (float)N_B), 0), N_B - 1);
        float el[DIM];
        const float* u = Eu + (long long)uids[b] * DIM;
        const float* p = Ev + (long long)pos[b]  * DIM;
#pragma unroll
        for (int d = 0; d < DIM; ++d)
            el[d] = 1.0f / (1.0f + __expf(-u[d] * p[d]));
        for (int k = 0; k < N_B; ++k) {
            float dot = 0.0f;
            const float* eb = Eb + k * DIM;
#pragma unroll
            for (int d = 0; d < DIM; ++d) dot += el[d] * eb[d];
            if (k == rel) tpos = dot; else tneg += dot;
        }
        tneg *= (1.0f / (float)N_B);
    }
    sn[threadIdx.x] = tneg; sp[threadIdx.x] = tpos;
    __syncthreads();
    for (int s = blockDim.x >> 1; s > 0; s >>= 1) {
        if (threadIdx.x < s) {
            sn[threadIdx.x] += sn[threadIdx.x + s];
            sp[threadIdx.x] += sp[threadIdx.x + s];
        }
        __syncthreads();
    }
    if (threadIdx.x == 0) {
        atomicAdd(&scal[S_BCLN], sn[0]);
        atomicAdd(&scal[S_BCLP], sp[0]);
    }
}

// ---------------------------------------------------------------------------
// sum of squares (L2 reg) and log-sum reductions
// ---------------------------------------------------------------------------
__global__ void k_sumsq(const float* __restrict__ x, float* __restrict__ scal, int n)
{
    __shared__ float sm[256];
    float s = 0.0f;
    for (long long i = (long long)blockIdx.x * blockDim.x + threadIdx.x; i < n;
         i += (long long)gridDim.x * blockDim.x) {
        float v = x[i]; s += v * v;
    }
    sm[threadIdx.x] = s;
    __syncthreads();
    for (int t = blockDim.x >> 1; t > 0; t >>= 1) {
        if (threadIdx.x < t) sm[threadIdx.x] += sm[threadIdx.x + t];
        __syncthreads();
    }
    if (threadIdx.x == 0) atomicAdd(&scal[S_REG], sm[0]);
}

__global__ void k_logsum(const float* __restrict__ acc, float* __restrict__ scal,
                         int slot, int batch)
{
    __shared__ float sm[256];
    float s = 0.0f;
    for (int i = threadIdx.x; i < batch; i += blockDim.x)
        s += __logf(acc[i] + 1e-8f);
    sm[threadIdx.x] = s;
    __syncthreads();
    for (int t = blockDim.x >> 1; t > 0; t >>= 1) {
        if (threadIdx.x < t) sm[threadIdx.x] += sm[threadIdx.x + t];
        __syncthreads();
    }
    if (threadIdx.x == 0) scal[slot] = sm[0];
}

__global__ void k_finalize(const float* __restrict__ scal, float* __restrict__ out)
{
    const float invB = 1.0f / (float)BATCH;
    float bpr = scal[S_BPR] * invB;
    float neg_s = scal[S_NEG_U] * invB + scal[S_NEG_I] * invB;
    float pos_s = scal[S_POS_U] * invB + scal[S_POS_I] * invB;
    float pcl = neg_s - pos_s;
    float bcl = scal[S_BCLN] * invB - scal[S_BCLP] * invB;
    float reg = LAMBDA_3 * scal[S_REG];
    out[0] = bpr + LAMBDA_1 * pcl + LAMBDA_2 * bcl + reg;
    out[1] = bpr;
    out[2] = LAMBDA_1 * pcl;
    out[3] = LAMBDA_2 * bcl;
}

// ---------------------------------------------------------------------------
// host orchestration
// ---------------------------------------------------------------------------
static inline int cdiv(long long n, int b) { return (int)((n + b - 1) / b); }

extern "C" void kernel_launch(void* const* d_in, const int* in_sizes, int n_in,
                              void* d_out, int out_size, void* d_ws, size_t ws_size,
                              hipStream_t stream)
{
    const float* Eu0  = (const float*)d_in[0];
    const float* Ev0  = (const float*)d_in[1];
    const float* Eb   = (const float*)d_in[2];
    const float* vals = (const float*)d_in[3];
    const int*   rows = (const int*)d_in[4];
    const int*   cols = (const int*)d_in[5];
    const int*   uids = (const int*)d_in[6];
    const int*   iids = (const int*)d_in[7];
    const int*   pos  = (const int*)d_in[8];
    const int*   neg  = (const int*)d_in[9];
    float* out = (float*)d_out;

    // carve workspace
    char* ws = (char*)d_ws;
    size_t off = 0;
    auto alloc = [&](size_t bytes) -> void* {
        off = (off + 255) & ~(size_t)255;
        void* p = ws + off;
        off += bytes;
        return p;
    };
    const size_t NU_F = (size_t)N_U * DIM * sizeof(float);
    const size_t NI_F = (size_t)N_I * DIM * sizeof(float);

    float* Eu_a  = (float*)alloc(NU_F);
    float* Eu_b  = (float*)alloc(NU_F);
    float* Ev_a  = (float*)alloc(NI_F);
    float* Ev_b  = (float*)alloc(NI_F);
    float* E_u   = (float*)alloc(NU_F);   // base-view sums
    float* E_v   = (float*)alloc(NI_F);
    float* Z_u   = (float*)alloc(NU_F);   // aug-view sums
    float* Z_i   = (float*)alloc(NI_F);
    float* augv  = (float*)alloc((size_t)N_EDGES * sizeof(float));
    unsigned int* Ebf_u = (unsigned int*)alloc((size_t)N_U * DIM * 2);
    unsigned int* Ebf_v = (unsigned int*)alloc((size_t)N_I * DIM * 2);
    unsigned int* Zbf_u = (unsigned int*)alloc((size_t)BATCH * DIM * 2);
    unsigned int* Zbf_i = (unsigned int*)alloc((size_t)BATCH * DIM * 2);
    float* ps      = (float*)alloc(BATCH * sizeof(float));
    float* negaccU = (float*)alloc(BATCH * sizeof(float));
    float* negaccI = (float*)alloc(BATCH * sizeof(float));
    float* scal    = (float*)alloc(NSCAL * sizeof(float));
    (void)ws_size; (void)in_sizes; (void)n_in; (void)out_size;

    const int NUe = N_U * DIM, NIe = N_I * DIM;

    k_zero<<<cdiv(NSCAL, 256), 256, 0, stream>>>(scal, NSCAL);
    k_zero<<<cdiv(BATCH, 256), 256, 0, stream>>>(negaccU, BATCH);
    k_zero<<<cdiv(BATCH, 256), 256, 0, stream>>>(negaccI, BATCH);

    // ---- two propagations: base (-> E_u/E_v) then augmented (-> Z_u/Z_i) ----
    for (int prop = 0; prop < 2; ++prop) {
        const float* pv = (prop == 0) ? vals : augv;
        float* Us = (prop == 0) ? E_u : Z_u;
        float* Vs = (prop == 0) ? E_v : Z_i;
        hipMemcpyAsync(Eu_a, Eu0, NU_F, hipMemcpyDeviceToDevice, stream);
        hipMemcpyAsync(Ev_a, Ev0, NI_F, hipMemcpyDeviceToDevice, stream);
        hipMemcpyAsync(Us,   Eu0, NU_F, hipMemcpyDeviceToDevice, stream);
        hipMemcpyAsync(Vs,   Ev0, NI_F, hipMemcpyDeviceToDevice, stream);
        float *ua = Eu_a, *ub = Eu_b, *va = Ev_a, *vb = Ev_b;
        for (int layer = 0; layer < NLAYER; ++layer) {
            unsigned seed = (unsigned)(prop * 100 + 2 * layer);
            k_zero<<<cdiv(NUe, 256), 256, 0, stream>>>(ub, NUe);
            k_zero<<<cdiv(NIe, 256), 256, 0, stream>>>(vb, NIe);
            // Eu_new = segsum(v1 * Ev[cols] -> rows)
            k_spmm<<<cdiv((long long)N_EDGES * 16, 256), 256, 0, stream>>>(
                pv, cols, rows, va, ub, seed, N_EDGES);
            // Ev_new = segsum(v2 * Eu[rows] -> cols)
            k_spmm<<<cdiv((long long)N_EDGES * 16, 256), 256, 0, stream>>>(
                pv, rows, cols, ua, vb, seed + 1, N_EDGES);
            k_add<<<cdiv(NUe, 256), 256, 0, stream>>>(Us, ub, NUe);
            k_add<<<cdiv(NIe, 256), 256, 0, stream>>>(Vs, vb, NIe);
            float* t;
            t = ua; ua = ub; ub = t;
            t = va; va = vb; vb = t;
        }
        if (prop == 0) {
            // learned edge weights -> augmented graph values
            k_edge_logits<<<cdiv(N_EDGES, 256), 256, 0, stream>>>(
                E_u, E_v, vals, rows, cols, augv, N_EDGES);
        }
    }

    // ---- BPR + bucket stats ----
    k_bpr<<<cdiv(BATCH, 256), 256, 0, stream>>>(E_u, E_v, uids, pos, neg, ps, scal, BATCH);
    k_minmax<<<1, 256, 0, stream>>>(ps, scal, BATCH);
    k_bcl<<<cdiv(BATCH, 256), 256, 0, stream>>>(E_u, E_v, Eb, uids, pos, ps, scal, BATCH);

    // ---- PCL positive (f32) ----
    k_pcl_pos<<<cdiv(BATCH, 256), 256, 0, stream>>>(Z_u, E_u, uids, scal, S_POS_U, BATCH);
    k_pcl_pos<<<cdiv(BATCH, 256), 256, 0, stream>>>(Z_i, E_v, iids, scal, S_POS_I, BATCH);

    // ---- PCL negative (bf16 WMMA logsumexp) ----
    k_conv_bf16<<<cdiv(NUe / 2, 256), 256, 0, stream>>>((const float2*)E_u, Ebf_u, NUe / 2);
    k_conv_bf16<<<cdiv(NIe / 2, 256), 256, 0, stream>>>((const float2*)E_v, Ebf_v, NIe / 2);
    k_gather_bf16<<<cdiv(BATCH * DIM / 2, 256), 256, 0, stream>>>(Z_u, uids, Zbf_u,
                                                                  1.0f / TEMP, BATCH);
    k_gather_bf16<<<cdiv(BATCH * DIM / 2, 256), 256, 0, stream>>>(Z_i, iids, Zbf_i,
                                                                  1.0f / TEMP, BATCH);
    {
        dim3 grid(BATCH / 16, 32);
        k_pcl_neg_wmma<<<grid, 256, 0, stream>>>(Zbf_u, Ebf_u, negaccU, N_U);
        k_pcl_neg_wmma<<<grid, 256, 0, stream>>>(Zbf_i, Ebf_v, negaccI, N_I);
    }
    k_logsum<<<1, 256, 0, stream>>>(negaccU, scal, S_NEG_U, BATCH);
    k_logsum<<<1, 256, 0, stream>>>(negaccI, scal, S_NEG_I, BATCH);

    // ---- L2 reg over parameters ----
    k_sumsq<<<4096, 256, 0, stream>>>(Eu0, scal, NUe);
    k_sumsq<<<4096, 256, 0, stream>>>(Ev0, scal, NIe);
    k_sumsq<<<4, 256, 0, stream>>>(Eb, scal, N_B * DIM);

    k_finalize<<<1, 1, 0, stream>>>(scal, out);
}
